// MPPCA_13357348291060
// MI455X (gfx1250) — matrix operand order repaired and checked by reading, not compiled, not run on hardware
//
#include <hip/hip_runtime.h>
#include <math.h>

// ---------------------------------------------------------------------------
// MPPCA log_prob for MI455X (gfx1250), wave32 + WMMA bf16.
// N=32768, D=512, K=64, L=16.
// ---------------------------------------------------------------------------

typedef __attribute__((ext_vector_type(16))) __bf16 v16bf;
typedef __attribute__((ext_vector_type(8)))  float  v8f;

#define NN 32768
#define DD 512
#define KK 64
#define LL 16

// round-to-nearest-even float -> bf16 bits
__device__ __forceinline__ unsigned short f2bf(float f) {
    unsigned int u = __float_as_uint(f);
    unsigned int r = (u + 0x7FFFu + ((u >> 16) & 1u)) >> 16;
    return (unsigned short)r;
}

// Load a 16-element bf16 B-fragment: lane holds N column (lane&15); p points at
// K-row 8*(lane>>4); elements K = 0..7 and 16..23 relative to p, row stride 'stride'.
__device__ __forceinline__ v16bf loadB32B(const unsigned short* p, int stride) {
    union { unsigned short h[16]; v16bf v; } f;
    #pragma unroll
    for (int s = 0; s < 8; ++s) {
        f.h[s]     = p[s * stride];
        f.h[8 + s] = p[(16 + s) * stride];
    }
    return f.v;
}

// ---------------------------------------------------------------------------
// Precompute kernel: one block per component k.
//  - WtW, Lmat = I + (1/s2) WtW, Cholesky G, logdet, Ginv
//  - W' = W * G^-T  (bf16, row-major [k][d][16])
//  - MuTp[d][k] = mu[k][d] (bf16), b[k][l] = sum_d W'[d][l]*mu[d]
//  - consts: inv_s2, musq, constk
// ---------------------------------------------------------------------------
__global__ void __launch_bounds__(256) mppca_prep(
    const float* __restrict__ pi, const float* __restrict__ mu,
    const float* __restrict__ W,  const float* __restrict__ sigma2,
    unsigned short* __restrict__ Wp, unsigned short* __restrict__ MuTp,
    float* __restrict__ bvec, float* __restrict__ inv_s2_o,
    float* __restrict__ musq_o, float* __restrict__ constk_o)
{
    const int k = blockIdx.x;
    const int t = threadIdx.x;

    __shared__ float sWk[DD * LL];      // 32 KB: W_k staged
    __shared__ float sMat[LL][LL + 1];  // Lmat
    __shared__ float sU[LL][LL + 1];    // U[m][l] = Ginv[l][m]
    __shared__ float sb[LL];
    __shared__ float sMusq;
    __shared__ float sLogdet;

    const float* Wk = W + (size_t)k * DD * LL;
    for (int i = t; i < DD * LL / 4; i += 256)
        ((float4*)sWk)[i] = ((const float4*)Wk)[i];
    if (t < LL) sb[t] = 0.0f;
    if (t == 0) sMusq = 0.0f;
    __syncthreads();

    const float is2 = 1.0f / sigma2[k];

    { // WtW entry per thread (256 = 16x16)
        const int i = t >> 4, j = t & 15;
        float s = 0.0f;
        for (int d = 0; d < DD; ++d) s += sWk[d * LL + i] * sWk[d * LL + j];
        sMat[i][j] = ((i == j) ? 1.0f : 0.0f) + is2 * s;
    }
    __syncthreads();

    if (t == 0) { // serial 16x16 Cholesky + triangular inverse (tiny)
        float G[LL][LL];
        float Gi[LL][LL];
        for (int i = 0; i < LL; ++i)
            for (int j = 0; j < LL; ++j) { G[i][j] = 0.0f; Gi[i][j] = 0.0f; }
        float logdet = 0.0f;
        for (int j = 0; j < LL; ++j) {
            float s = sMat[j][j];
            for (int m = 0; m < j; ++m) s -= G[j][m] * G[j][m];
            float gjj = sqrtf(s);
            G[j][j] = gjj;
            logdet += 2.0f * logf(gjj);
            float inv = 1.0f / gjj;
            for (int i = j + 1; i < LL; ++i) {
                float v = sMat[i][j];
                for (int m = 0; m < j; ++m) v -= G[i][m] * G[j][m];
                G[i][j] = v * inv;
            }
        }
        sLogdet = logdet;
        for (int j = 0; j < LL; ++j) {
            Gi[j][j] = 1.0f / G[j][j];
            for (int i = j + 1; i < LL; ++i) {
                float s = 0.0f;
                for (int m = j; m < i; ++m) s += G[i][m] * Gi[m][j];
                Gi[i][j] = -s / G[i][i];
            }
        }
        // U = Ginv^T  (invL = U U^T), nonzero for m <= l
        for (int m = 0; m < LL; ++m)
            for (int l = 0; l < LL; ++l) sU[m][l] = Gi[l][m];
    }
    __syncthreads();

    const float* muk = mu + (size_t)k * DD;
    float bloc[LL];
    for (int l = 0; l < LL; ++l) bloc[l] = 0.0f;
    float msq = 0.0f;
    for (int d = t; d < DD; d += 256) {
        float wr[LL];
        for (int m = 0; m < LL; ++m) wr[m] = sWk[d * LL + m];
        const float mud = muk[d];
        msq += mud * mud;
        unsigned short* op = Wp + ((size_t)k * DD + d) * LL;
        for (int l = 0; l < LL; ++l) {
            float s = 0.0f;
            for (int m = 0; m <= l; ++m) s += wr[m] * sU[m][l];
            op[l] = f2bf(s);
            bloc[l] += s * mud;
        }
        MuTp[d * KK + k] = f2bf(mud);
    }
    for (int l = 0; l < LL; ++l) atomicAdd(&sb[l], bloc[l]);
    atomicAdd(&sMusq, msq);
    __syncthreads();

    if (t < LL) bvec[k * LL + t] = sb[t];
    if (t == 0) {
        inv_s2_o[k] = is2;
        musq_o[k]   = sMusq;
        const float LOG2PI = 1.8378770664093453f;
        constk_o[k] = logf(pi[k])
                    - 0.5f * ((float)DD * LOG2PI + sLogdet + (float)DD * logf(sigma2[k]));
    }
}

// ---------------------------------------------------------------------------
// Main kernel: 64 rows per block, 256 threads (8 wave32).
// Waves 0-3 own 16-row strips with k in [0,32); waves 4-7 same strips, k [32,64).
// ---------------------------------------------------------------------------
#define ROWS   64
#define XPITCH 520   // bf16 elements per row (512 + 8 pad -> conflict-free b128)

__global__ void __launch_bounds__(256) mppca_main(
    const float* __restrict__ x,
    const unsigned short* __restrict__ Wp,
    const unsigned short* __restrict__ MuTp,
    const float* __restrict__ bvec,
    const float* __restrict__ inv_s2,
    const float* __restrict__ musq,
    const float* __restrict__ constk,
    float* __restrict__ out)
{
    __shared__ unsigned short sX[ROWS * XPITCH]; // 66,560 B bf16 x-tile
    __shared__ float sXmu[ROWS][KK + 1];         // x . mu_k
    __shared__ float sQuad[ROWS][KK + 1];        // ||W'^T x - b||^2
    __shared__ float sXsq[ROWS];

    const int t    = threadIdx.x;
    const int row0 = blockIdx.x * ROWS;

    if (t < ROWS) sXsq[t] = 0.0f;
    __syncthreads();

    // ---- Stage 1: load x tile, convert to bf16 in LDS, accumulate ||x||^2
    {
        const int r    = t >> 2;          // 4 threads per row
        const int cseg = (t & 3) * 128;
        const float* xr = x + (size_t)(row0 + r) * DD + cseg;
        float acc = 0.0f;
        #pragma unroll 8
        for (int c = 0; c < 128; c += 4) {
            float4 v = *(const float4*)(xr + c);
            acc += v.x * v.x + v.y * v.y + v.z * v.z + v.w * v.w;
            unsigned int p0 = (unsigned int)f2bf(v.x) | ((unsigned int)f2bf(v.y) << 16);
            unsigned int p1 = (unsigned int)f2bf(v.z) | ((unsigned int)f2bf(v.w) << 16);
            *(uint2*)&sX[r * XPITCH + cseg + c] = make_uint2(p0, p1);
        }
        atomicAdd(&sXsq[r], acc);
    }
    __syncthreads();

    const int wave  = t >> 5;
    const int lane  = t & 31;
    const int strip = wave & 3;   // 16-row strip
    const int khalf = wave >> 2;  // component half
    const int laneM = lane & 15;
    const int laneH = lane >> 4;
    const int rowA  = strip * 16 + laneM;
    const int kb    = laneH * 8;                 // A-fragment K interleave
    const int dlane = laneM + laneH * 16;        // B-fragment K row in 32-block

    // ---- Hoist all 16 A-fragments (16-bit A 16x32 layout) into registers
    union FragU { unsigned int u[8]; v16bf v; } aF[16];
    {
        const unsigned short* arow = &sX[rowA * XPITCH];
        #pragma unroll
        for (int kk = 0; kk < 16; ++kk) {
            const int c = kk * 32 + kb;
            uint4 a = *(const uint4*)(arow + c);        // K = kb .. kb+7
            uint4 b = *(const uint4*)(arow + c + 16);   // K = kb+16 .. kb+23
            aF[kk].u[0]=a.x; aF[kk].u[1]=a.y; aF[kk].u[2]=a.z; aF[kk].u[3]=a.w;
            aF[kk].u[4]=b.x; aF[kk].u[5]=b.y; aF[kk].u[6]=b.z; aF[kk].u[7]=b.w;
        }
    }

    // ---- Stage 2a: xmu = X * mu^T for this wave's 2 blocks of 16 components
    {
        v8f acc0 = {0.f,0.f,0.f,0.f,0.f,0.f,0.f,0.f};
        v8f acc1 = {0.f,0.f,0.f,0.f,0.f,0.f,0.f,0.f};
        const int blk0 = khalf * 2, blk1 = khalf * 2 + 1;
        #pragma unroll
        for (int kk = 0; kk < 16; ++kk) {
            const int dRow = kk * 32 + kb;
            v16bf b0 = loadB32B(MuTp + dRow * KK + blk0 * 16 + laneM, KK);
            v16bf b1 = loadB32B(MuTp + dRow * KK + blk1 * 16 + laneM, KK);
            acc0 = __builtin_amdgcn_wmma_f32_16x16x32_bf16(
                       false, aF[kk].v, false, b0, (short)0, acc0, false, false);
            acc1 = __builtin_amdgcn_wmma_f32_16x16x32_bf16(
                       false, aF[kk].v, false, b1, (short)0, acc1, false, false);
        }
        const int rowD = strip * 16 + laneH * 8;   // D-frag: VGPR r -> M = r + 8*laneH
        #pragma unroll
        for (int r = 0; r < 8; ++r) {
            sXmu[rowD + r][blk0 * 16 + laneM] = acc0[r];
            sXmu[rowD + r][blk1 * 16 + laneM] = acc1[r];
        }
    }

    // ---- Stage 2b: per component, z = W'^T x - b ; quad = ||z||^2 (row-sum)
    #pragma unroll 2
    for (int k = khalf * 32; k < khalf * 32 + 32; ++k) {
        v8f acc = {0.f,0.f,0.f,0.f,0.f,0.f,0.f,0.f};
        const unsigned short* wk = Wp + (size_t)k * DD * LL;
        #pragma unroll
        for (int kk = 0; kk < 16; ++kk) {
            const int dRow = kk * 32 + kb;
            v16bf b = loadB32B(wk + dRow * LL + laneM, LL);
            acc = __builtin_amdgcn_wmma_f32_16x16x32_bf16(
                      false, aF[kk].v, false, b, (short)0, acc, false, false);
        }
        const float bk = bvec[k * LL + laneM];     // column (L-dim) offset
        float sum[8];
        #pragma unroll
        for (int r = 0; r < 8; ++r) { float z = acc[r] - bk; sum[r] = z * z; }
        // reduce across the 16 lanes of each half (columns of the L dim)
        #pragma unroll
        for (int m = 1; m <= 8; m <<= 1) {
            #pragma unroll
            for (int r = 0; r < 8; ++r) sum[r] += __shfl_xor(sum[r], m, 32);
        }
        if (laneM == 0) {
            const int rowD = strip * 16 + laneH * 8;
            #pragma unroll
            for (int r = 0; r < 8; ++r) sQuad[rowD + r][k] = sum[r];
        }
    }
    __syncthreads();

    // ---- Stage 3: per-row logsumexp over K components
    if (t < ROWS) {
        const int row = t;
        const float xsq = sXsq[row];
        float mmax = -1e30f, ssum = 0.0f;
        for (int k = 0; k < KK; ++k) {
            const float is2 = inv_s2[k];
            const float sq  = xsq - 2.0f * sXmu[row][k] + musq[k];
            const float md  = is2 * sq - is2 * is2 * sQuad[row][k];
            const float ll  = constk[k] - 0.5f * md;
            if (ll > mmax) { ssum = ssum * expf(mmax - ll) + 1.0f; mmax = ll; }
            else           { ssum += expf(ll - mmax); }
        }
        out[row0 + row] = mmax + logf(ssum);
    }
}

// ---------------------------------------------------------------------------
extern "C" void kernel_launch(void* const* d_in, const int* in_sizes, int n_in,
                              void* d_out, int out_size, void* d_ws, size_t ws_size,
                              hipStream_t stream) {
    const float* x      = (const float*)d_in[0];
    const float* pi     = (const float*)d_in[1];
    const float* mu     = (const float*)d_in[2];
    const float* W      = (const float*)d_in[3];
    const float* sigma2 = (const float*)d_in[4];
    float* out = (float*)d_out;

    char* ws = (char*)d_ws;
    unsigned short* Wp    = (unsigned short*)(ws);                 // 1,048,576 B
    unsigned short* MuTp  = (unsigned short*)(ws + 1048576);       //    65,536 B
    float*          bvec  = (float*)(ws + 1114112);                //     4,096 B
    float*          is2   = (float*)(ws + 1118208);                //       256 B
    float*          musq  = (float*)(ws + 1118464);                //       256 B
    float*          ck    = (float*)(ws + 1118720);                //       256 B

    mppca_prep<<<KK, 256, 0, stream>>>(pi, mu, W, sigma2, Wp, MuTp, bvec, is2, musq, ck);
    mppca_main<<<NN / ROWS, 256, 0, stream>>>(x, Wp, MuTp, bvec, is2, musq, ck, out);
}
